// Block_54477365182485
// MI455X (gfx1250) — matrix operand order repaired and checked by reading, not compiled, not run on hardware
//
#include <hip/hip_runtime.h>
#include <hip/hip_bf16.h>
#include <cstdint>
#include <cstddef>

// ---------------------------------------------------------------------------
// Types for CDNA5 WMMA
// ---------------------------------------------------------------------------
typedef __bf16 bf16;
typedef __attribute__((ext_vector_type(16))) __bf16 v16bf;
typedef __attribute__((ext_vector_type(8)))  __bf16 v8bf;
typedef __attribute__((ext_vector_type(4)))  __bf16 v4bf;
typedef __attribute__((ext_vector_type(8)))  float  v8f;

union V16U { v16bf v; struct { v8bf lo; v8bf hi; } h; };

// CDNA5 async global->LDS path (ASYNCcnt), if the toolchain exposes it.
#if defined(__AMDGCN__) && defined(__has_builtin)
#if __has_builtin(__builtin_amdgcn_global_load_async_to_lds_b128) && \
    __has_builtin(__builtin_amdgcn_s_wait_asynccnt)
#define USE_ASYNC_LDS 1
#endif
#endif
#ifndef USE_ASYNC_LDS
#define USE_ASYNC_LDS 0
#endif

#if USE_ASYNC_LDS
typedef __attribute__((ext_vector_type(4))) int v4i;
typedef __attribute__((address_space(1))) v4i as1_v4i;   // global
typedef __attribute__((address_space(3))) v4i as3_v4i;   // LDS
#endif

// Problem constants (match reference)
constexpr int CB   = 8;
constexpr int CL   = 2048;
constexpr int CDIM = 1024;
constexpr int CH   = 16;
constexpr int CD   = 64;
constexpr int CHID = 4096;
constexpr int CU   = 40;          // u == u_part == 40
constexpr int CM   = CB * CL;     // 16384 rows of activations

// ---------------------------------------------------------------------------
// fp32 -> bf16 conversion (vectorized x4)
// ---------------------------------------------------------------------------
__global__ void cvt_kernel(const float* __restrict__ x, bf16* __restrict__ y,
                           size_t n4) {
  size_t i = (size_t)blockIdx.x * blockDim.x + threadIdx.x;
  if (i >= n4) return;
  float4 f = ((const float4*)x)[i];
  v4bf o;
  o.x = (__bf16)f.x; o.y = (__bf16)f.y; o.z = (__bf16)f.z; o.w = (__bf16)f.w;
  ((v4bf*)y)[i] = o;
}

// ---------------------------------------------------------------------------
// bf16 WMMA GEMM:  C[M,N] = A[M,K] * W[N,K]^T + bias[N]   (fp32 out)
// Block: 256 thr = 8 waves. Block tile 128(M) x 128(N), wave tile 32x64
// (2 A-frags x 4 B-frags = 8 WMMA per K-step of 32). A/B tiles staged in
// LDS with double buffering; async global->LDS when available.
// Requires M%128==0, N%128==0, K%32==0 (true for all call sites).
// ---------------------------------------------------------------------------
__device__ inline v16bf load_a_frag(const bf16* p) {
  // lane m (0..15): K = 0..15 ; lane m+16: K = 16..31
  // caller pre-offsets p by half*16; we read p[0..15]
  V16U u;
  u.h.lo = *(const v8bf*)(p);
  u.h.hi = *(const v8bf*)(p + 8);
  return u.v;
}

// copy a 128x32 bf16 tile (row stride ldsrc) into LDS (row stride 32)
// 256 threads * 2 iterations * 16B = 8KB ; 2 async instructions per wave
__device__ inline void stage128x32(const bf16* __restrict__ src, size_t ldsrc,
                                   bf16* lds, int tid) {
#pragma unroll
  for (int i = 0; i < 2; ++i) {
    const int e   = tid + i * 256;      // 0..511
    const int row = e >> 2;
    const int seg = e & 3;              // 16B segment within the 64B row
    const bf16* gp = src + (size_t)row * ldsrc + seg * 8;
    bf16* lp = lds + row * 32 + seg * 8;
#if USE_ASYNC_LDS
    __builtin_amdgcn_global_load_async_to_lds_b128(
        (as1_v4i*)gp, (as3_v4i*)lp, 0, 0);
#else
    *(v8bf*)lp = *(const v8bf*)gp;
#endif
  }
}

__global__ __launch_bounds__(256)
void gemm_bf16_wmma(const bf16* __restrict__ A, const bf16* __restrict__ W,
                    const float* __restrict__ bias, float* __restrict__ C,
                    int M, int N, int K) {
  __shared__ bf16 lA[2][128 * 32];
  __shared__ bf16 lB[2][128 * 32];

  const int tid  = threadIdx.x;
  const int lane = tid & 31;
  const int wave = tid >> 5;
  const int wm   = wave & 3;             // 0..3  -> 32-row subtile
  const int wn   = wave >> 2;            // 0..1  -> 64-col subtile
  const int half = lane >> 4;            // 0/1
  const int l16  = lane & 15;

  const int mBlk = blockIdx.y * 128;
  const int nBlk = blockIdx.x * 128;

  v8f acc[2][4];
#pragma unroll
  for (int i = 0; i < 2; ++i)
#pragma unroll
    for (int j = 0; j < 4; ++j) acc[i][j] = (v8f){};

  // prologue: stage tile 0 into buffer 0
  stage128x32(A + (size_t)mBlk * K, K, lA[0], tid);
  stage128x32(W + (size_t)nBlk * K, K, lB[0], tid);

  int buf = 0;
  for (int k = 0; k < K; k += 32) {
    const int nk = k + 32;
    if (nk < K) {
      stage128x32(A + (size_t)mBlk * K + nk, K, lA[buf ^ 1], tid);
      stage128x32(W + (size_t)nBlk * K + nk, K, lB[buf ^ 1], tid);
#if USE_ASYNC_LDS
      __builtin_amdgcn_s_wait_asynccnt(4);   // retire current tile's 4 ops
#endif
    } else {
#if USE_ASYNC_LDS
      __builtin_amdgcn_s_wait_asynccnt(0);
#endif
    }
    __syncthreads();

    const bf16* la = lA[buf];
    const bf16* lb = lB[buf];

    v16bf a0 = load_a_frag(la + (wm * 32 + l16) * 32 + half * 16);
    v16bf a1 = load_a_frag(la + (wm * 32 + 16 + l16) * 32 + half * 16);
    v16bf b0 = *(const v16bf*)(lb + (wn * 64 + 0  + l16) * 32 + half * 16);
    v16bf b1 = *(const v16bf*)(lb + (wn * 64 + 16 + l16) * 32 + half * 16);
    v16bf b2 = *(const v16bf*)(lb + (wn * 64 + 32 + l16) * 32 + half * 16);
    v16bf b3 = *(const v16bf*)(lb + (wn * 64 + 48 + l16) * 32 + half * 16);

    acc[0][0] = __builtin_amdgcn_wmma_f32_16x16x32_bf16(false, a0, false, b0,
                                                        (short)0, acc[0][0], false, false);
    acc[0][1] = __builtin_amdgcn_wmma_f32_16x16x32_bf16(false, a0, false, b1,
                                                        (short)0, acc[0][1], false, false);
    acc[0][2] = __builtin_amdgcn_wmma_f32_16x16x32_bf16(false, a0, false, b2,
                                                        (short)0, acc[0][2], false, false);
    acc[0][3] = __builtin_amdgcn_wmma_f32_16x16x32_bf16(false, a0, false, b3,
                                                        (short)0, acc[0][3], false, false);
    acc[1][0] = __builtin_amdgcn_wmma_f32_16x16x32_bf16(false, a1, false, b0,
                                                        (short)0, acc[1][0], false, false);
    acc[1][1] = __builtin_amdgcn_wmma_f32_16x16x32_bf16(false, a1, false, b1,
                                                        (short)0, acc[1][1], false, false);
    acc[1][2] = __builtin_amdgcn_wmma_f32_16x16x32_bf16(false, a1, false, b2,
                                                        (short)0, acc[1][2], false, false);
    acc[1][3] = __builtin_amdgcn_wmma_f32_16x16x32_bf16(false, a1, false, b3,
                                                        (short)0, acc[1][3], false, false);

    __syncthreads();   // all waves done reading before buffer is re-armed
    buf ^= 1;
  }

  // C/D layout: VGPR r -> row (r + 8*half), col l16 within each 16x16 tile
  const int m0 = mBlk + wm * 32 + half * 8;
#pragma unroll
  for (int i = 0; i < 2; ++i) {
#pragma unroll
    for (int j = 0; j < 4; ++j) {
      const int col = nBlk + wn * 64 + j * 16 + l16;
      const float bj = bias ? bias[col] : 0.f;
      float* crow = C + (size_t)(m0 + i * 16) * N + col;
#pragma unroll
      for (int r = 0; r < 8; ++r) crow[(size_t)r * N] = acc[i][j][r] + bj;
    }
  }
}

// ---------------------------------------------------------------------------
// LayerNorm over rows (optionally fused exact GELU). One block per row.
// ---------------------------------------------------------------------------
__global__ __launch_bounds__(256)
void ln_kernel(const float* __restrict__ x, const float* __restrict__ w,
               const float* __restrict__ b, float* __restrict__ out,
               int cols, int do_gelu) {
  __shared__ float s_sum[256];
  __shared__ float s_sq[256];
  const int tid = threadIdx.x;
  const size_t row = blockIdx.x;
  const float* xr = x + row * cols;
  float* orow = out + row * cols;

  float sum = 0.f, sq = 0.f;
  for (int i = tid; i < cols; i += 256) { float v = xr[i]; sum += v; sq += v * v; }
  s_sum[tid] = sum; s_sq[tid] = sq; __syncthreads();
  for (int s = 128; s > 0; s >>= 1) {
    if (tid < s) { s_sum[tid] += s_sum[tid + s]; s_sq[tid] += s_sq[tid + s]; }
    __syncthreads();
  }
  const float mean = s_sum[0] / (float)cols;
  const float var  = s_sq[0] / (float)cols - mean * mean;
  const float inv  = rsqrtf(var + 1e-5f);
  for (int i = tid; i < cols; i += 256) {
    float v = (xr[i] - mean) * inv * w[i] + b[i];
    if (do_gelu) v = 0.5f * v * (1.f + erff(v * 0.70710678118654752f));
    orow[i] = v;
  }
}

// LayerNorm(x + y) — used for the post-attention residual
__global__ __launch_bounds__(256)
void add_ln_kernel(const float* __restrict__ x, const float* __restrict__ y,
                   const float* __restrict__ w, const float* __restrict__ b,
                   float* __restrict__ out, int cols) {
  __shared__ float s_sum[256];
  __shared__ float s_sq[256];
  const int tid = threadIdx.x;
  const size_t row = blockIdx.x;
  const float* xr = x + row * cols;
  const float* yr = y + row * cols;
  float* orow = out + row * cols;

  float sum = 0.f, sq = 0.f;
  for (int i = tid; i < cols; i += 256) {
    float v = xr[i] + yr[i]; sum += v; sq += v * v;
  }
  s_sum[tid] = sum; s_sq[tid] = sq; __syncthreads();
  for (int s = 128; s > 0; s >>= 1) {
    if (tid < s) { s_sum[tid] += s_sum[tid + s]; s_sq[tid] += s_sq[tid + s]; }
    __syncthreads();
  }
  const float mean = s_sum[0] / (float)cols;
  const float var  = s_sq[0] / (float)cols - mean * mean;
  const float inv  = rsqrtf(var + 1e-5f);
  for (int i = tid; i < cols; i += 256) {
    float v = xr[i] + yr[i];
    orow[i] = (v - mean) * inv * w[i] + b[i];
  }
}

__global__ void add_kernel(const float* __restrict__ a, const float* __restrict__ b,
                           float* __restrict__ out, size_t n) {
  size_t i = (size_t)blockIdx.x * blockDim.x + threadIdx.x;
  if (i < n) out[i] = a[i] + b[i];
}

// ---------------------------------------------------------------------------
// ProbSparse attention pieces. q,k,v stored as (B*L, DIM) with head h at
// columns [h*64, h*64+64).
// ---------------------------------------------------------------------------

// sparsity measure m[b,h,l] = max_s(q.k_sample) - sum_s(q.k_sample)/L
__global__ __launch_bounds__(64)
void qk_m_kernel(const float* __restrict__ q, const float* __restrict__ k,
                 const int* __restrict__ idx, float* __restrict__ m) {
  __shared__ float s[CU];
  const int gid = blockIdx.x;          // (b*H + h)*L + l
  const int l   = gid % CL;
  const int bh  = gid / CL;
  const int h   = bh % CH;
  const int b   = bh / CH;
  const int t   = threadIdx.x;
  if (t < CU) {
    const int si = idx[l * CU + t];
    const float* qp = q + ((size_t)(b * CL + l))  * CDIM + h * CD;
    const float* kp = k + ((size_t)(b * CL + si)) * CDIM + h * CD;
    float d = 0.f;
#pragma unroll 8
    for (int e = 0; e < CD; ++e) d += qp[e] * kp[e];
    s[t] = d;
  }
  __syncthreads();
  if (t == 0) {
    float mx = s[0], sm = 0.f;
    for (int i = 0; i < CU; ++i) { mx = fmaxf(mx, s[i]); sm += s[i]; }
    m[(size_t)bh * CL + l] = mx - sm / (float)CL;
  }
}

// iterative top-40 per (b,h): 40 rounds of block argmax over 2048 values
__global__ __launch_bounds__(256)
void topk_kernel(const float* __restrict__ m, int* __restrict__ mtop) {
  __shared__ float vals[CL];
  __shared__ float rv[256];
  __shared__ int   ri[256];
  const int bh = blockIdx.x;
  const int tid = threadIdx.x;
  for (int i = tid; i < CL; i += 256) vals[i] = m[(size_t)bh * CL + i];
  __syncthreads();
  for (int it = 0; it < CU; ++it) {
    float bv = -3.4e38f; int bi = 0;
    for (int i = tid; i < CL; i += 256)
      if (vals[i] > bv) { bv = vals[i]; bi = i; }
    rv[tid] = bv; ri[tid] = bi; __syncthreads();
    for (int s = 128; s > 0; s >>= 1) {
      if (tid < s && rv[tid + s] > rv[tid]) { rv[tid] = rv[tid + s]; ri[tid] = ri[tid + s]; }
      __syncthreads();
    }
    if (tid == 0) { mtop[bh * CU + it] = ri[0]; vals[ri[0]] = -3.4e38f; }
    __syncthreads();
  }
}

// softmax(q_sel . K / sqrt(d)) @ V for each selected row; one block per
// (b,h,u). Unnormalized exp accumulated with LDS float atomics (ds_add_f32).
__global__ __launch_bounds__(256)
void attn_update_kernel(const float* __restrict__ q, const float* __restrict__ k,
                        const float* __restrict__ v, const int* __restrict__ mtop,
                        float* __restrict__ upd_out) {
  __shared__ float sc[CL];     // scores / exp
  __shared__ float qs[CD];
  __shared__ float upd[CD];
  __shared__ float red[256];
  const int g  = blockIdx.x;       // bh*CU + u
  const int bh = g / CU;
  const int h  = bh % CH;
  const int b  = bh / CH;
  const int tid = threadIdx.x;

  const int lq = mtop[g];
  if (tid < CD) {
    qs[tid]  = q[((size_t)(b * CL + lq)) * CDIM + h * CD + tid];
    upd[tid] = 0.f;
  }
  __syncthreads();

  // scores
  float lmax = -3.4e38f;
#pragma unroll
  for (int j = 0; j < CL / 256; ++j) {
    const int kk = tid + j * 256;
    const float* kp = k + ((size_t)(b * CL + kk)) * CDIM + h * CD;
    float d = 0.f;
#pragma unroll 8
    for (int e = 0; e < CD; ++e) d += qs[e] * kp[e];
    d *= 0.125f;  // 1/sqrt(64)
    sc[kk] = d;
    lmax = fmaxf(lmax, d);
  }
  // block max
  red[tid] = lmax; __syncthreads();
  for (int s = 128; s > 0; s >>= 1) {
    if (tid < s) red[tid] = fmaxf(red[tid], red[tid + s]);
    __syncthreads();
  }
  const float mx = red[0]; __syncthreads();

  // exp + block sum
  float lsum = 0.f;
#pragma unroll
  for (int j = 0; j < CL / 256; ++j) {
    const int kk = tid + j * 256;
    float e = expf(sc[kk] - mx);
    sc[kk] = e;
    lsum += e;
  }
  red[tid] = lsum; __syncthreads();
  for (int s = 128; s > 0; s >>= 1) {
    if (tid < s) red[tid] += red[tid + s];
    __syncthreads();
  }
  const float S = red[0]; __syncthreads();

  // accumulate unnormalized attn @ V
  const float* vp[CL / 256];
#pragma unroll
  for (int j = 0; j < CL / 256; ++j) {
    const int kk = tid + j * 256;
    vp[j] = v + ((size_t)(b * CL + kk)) * CDIM + h * CD;
  }
  for (int d = 0; d < CD; ++d) {
    float acc = 0.f;
#pragma unroll
    for (int j = 0; j < CL / 256; ++j) acc += sc[tid + j * 256] * vp[j][d];
    atomicAdd(&upd[d], acc);   // ds_add_f32
  }
  __syncthreads();
  if (tid < CD) upd_out[(size_t)g * CD + tid] = upd[tid] / S;
}

// per-(b,h,d) mean of V over sequence
__global__ __launch_bounds__(64)
void vmean_kernel(const float* __restrict__ v, float* __restrict__ vmean) {
  const int bh = blockIdx.x;
  const int h  = bh % CH;
  const int b  = bh / CH;
  const int d  = threadIdx.x;
  float s = 0.f;
  const float* vp = v + ((size_t)(b * CL)) * CDIM + h * CD + d;
  for (int l = 0; l < CL; ++l) s += vp[(size_t)l * CDIM];
  vmean[(size_t)bh * CD + d] = s / (float)CL;
}

// ctx[b,l,h,d] = vmean[b,h,d] (broadcast fill)
__global__ void ctx_fill_kernel(const float* __restrict__ vmean,
                                float* __restrict__ ctx) {
  const size_t i = (size_t)blockIdx.x * blockDim.x + threadIdx.x;
  if (i >= (size_t)CM * CDIM) return;
  const int col = (int)(i % CDIM);
  const size_t row = i / CDIM;
  const int b = (int)(row / CL);
  const int h = col >> 6;
  const int d = col & 63;
  ctx[i] = vmean[((size_t)(b * CH + h)) * CD + d];
}

// scatter the updated rows into ctx
__global__ __launch_bounds__(64)
void ctx_scatter_kernel(const int* __restrict__ mtop,
                        const float* __restrict__ upd,
                        float* __restrict__ ctx) {
  const int g  = blockIdx.x;       // bh*CU + u
  const int bh = g / CU;
  const int h  = bh % CH;
  const int b  = bh / CH;
  const int lq = mtop[g];
  ctx[((size_t)(b * CL + lq)) * CDIM + h * CD + threadIdx.x] =
      upd[(size_t)g * CD + threadIdx.x];
}

// ---------------------------------------------------------------------------
// Host orchestration
// ---------------------------------------------------------------------------
extern "C" void kernel_launch(void* const* d_in, const int* in_sizes, int n_in,
                              void* d_out, int out_size, void* d_ws, size_t ws_size,
                              hipStream_t stream) {
  (void)in_sizes; (void)n_in; (void)out_size; (void)ws_size;

  const float* x_before = (const float*)d_in[0];
  const float* x_after  = (const float*)d_in[1];
  const float* norm1_w  = (const float*)d_in[2];
  const float* norm1_b  = (const float*)d_in[3];
  const float* wq       = (const float*)d_in[4];
  const float* bq       = (const float*)d_in[5];
  const float* wk       = (const float*)d_in[6];
  const float* bk       = (const float*)d_in[7];
  const float* wv       = (const float*)d_in[8];
  const float* bv       = (const float*)d_in[9];
  const float* wo       = (const float*)d_in[10];
  const float* bo       = (const float*)d_in[11];
  const float* norm_w   = (const float*)d_in[12];
  const float* norm_b   = (const float*)d_in[13];
  const float* fc1_w    = (const float*)d_in[14];
  const float* fc1_b    = (const float*)d_in[15];
  const float* ln1_w    = (const float*)d_in[16];
  const float* ln1_b    = (const float*)d_in[17];
  const float* fc2_w    = (const float*)d_in[18];
  const float* fc2_b    = (const float*)d_in[19];
  const float* ln2_w    = (const float*)d_in[20];
  const float* ln2_b    = (const float*)d_in[21];
  const int*   idx      = (const int*)d_in[22];
  float* out = (float*)d_out;

  // ---- workspace carve-up -------------------------------------------------
  char* ws = (char*)d_ws;
  size_t off = 0;
  auto alloc = [&](size_t bytes) -> char* {
    char* p = ws + off;
    off += (bytes + 255) & ~(size_t)255;
    return p;
  };
  bf16* wq16  = (bf16*)alloc((size_t)CDIM * CDIM * sizeof(bf16));
  bf16* wk16  = (bf16*)alloc((size_t)CDIM * CDIM * sizeof(bf16));
  bf16* wv16  = (bf16*)alloc((size_t)CDIM * CDIM * sizeof(bf16));
  bf16* wo16  = (bf16*)alloc((size_t)CDIM * CDIM * sizeof(bf16));
  bf16* fc1w16 = (bf16*)alloc((size_t)CHID * CDIM * sizeof(bf16));
  bf16* fc2w16 = (bf16*)alloc((size_t)CDIM * CHID * sizeof(bf16));
  bf16* act16 = (bf16*)alloc((size_t)CM * CHID * sizeof(bf16));   // biggest A
  float* nb     = (float*)alloc((size_t)CM * CDIM * sizeof(float)); // before_norm
  float* na     = (float*)alloc((size_t)CM * CDIM * sizeof(float)); // after_norm
  float* qbuf   = (float*)alloc((size_t)CM * CDIM * sizeof(float));
  float* kbuf   = (float*)alloc((size_t)CM * CDIM * sizeof(float));
  float* vbuf   = (float*)alloc((size_t)CM * CDIM * sizeof(float));
  float* ctx    = (float*)alloc((size_t)CM * CDIM * sizeof(float));
  float* tmp    = (float*)alloc((size_t)CM * CDIM * sizeof(float));
  float* hid    = (float*)alloc((size_t)CM * CHID * sizeof(float));
  float* bstate = (float*)alloc((size_t)CM * CDIM * sizeof(float));
  float* astate = (float*)alloc((size_t)CM * CDIM * sizeof(float));
  float* mbuf   = (float*)alloc((size_t)CB * CH * CL * sizeof(float));
  int*   mtop   = (int*)  alloc((size_t)CB * CH * CU * sizeof(int));
  float* upd    = (float*)alloc((size_t)CB * CH * CU * CD * sizeof(float));
  float* vmean  = (float*)alloc((size_t)CB * CH * CD * sizeof(float));

  // ---- helpers ------------------------------------------------------------
  auto cvt = [&](const float* src, bf16* dst, size_t n) {
    size_t n4 = n / 4;
    cvt_kernel<<<dim3((unsigned)((n4 + 255) / 256)), 256, 0, stream>>>(src, dst, n4);
  };
  auto gemm = [&](const bf16* A, const bf16* Wt, const float* bias, float* Co,
                  int Mm, int Nn, int Kk) {
    dim3 g(Nn / 128, Mm / 128);
    gemm_bf16_wmma<<<g, 256, 0, stream>>>(A, Wt, bias, Co, Mm, Nn, Kk);
  };
  auto ln = [&](const float* x, const float* w, const float* b, float* o,
                int cols, int gelu) {
    ln_kernel<<<dim3(CM), 256, 0, stream>>>(x, w, b, o, cols, gelu);
  };

  // ---- convert weights to bf16 -------------------------------------------
  cvt(wq, wq16, (size_t)CDIM * CDIM);
  cvt(wk, wk16, (size_t)CDIM * CDIM);
  cvt(wv, wv16, (size_t)CDIM * CDIM);
  cvt(wo, wo16, (size_t)CDIM * CDIM);
  cvt(fc1_w, fc1w16, (size_t)CHID * CDIM);
  cvt(fc2_w, fc2w16, (size_t)CDIM * CHID);

  // ---- input layer norms --------------------------------------------------
  ln(x_before, norm1_w, norm1_b, nb, CDIM, 0);
  ln(x_after,  norm1_w, norm1_b, na, CDIM, 0);

  // ---- ProbSparse attention (q from qsrc, k/v from kvsrc) -> tmp ----------
  auto attention = [&](const float* qsrc, const float* kvsrc) {
    cvt(qsrc, act16, (size_t)CM * CDIM);
    gemm(act16, wq16, bq, qbuf, CM, CDIM, CDIM);
    cvt(kvsrc, act16, (size_t)CM * CDIM);
    gemm(act16, wk16, bk, kbuf, CM, CDIM, CDIM);
    gemm(act16, wv16, bv, vbuf, CM, CDIM, CDIM);

    qk_m_kernel<<<dim3(CB * CH * CL), 64, 0, stream>>>(qbuf, kbuf, idx, mbuf);
    topk_kernel<<<dim3(CB * CH), 256, 0, stream>>>(mbuf, mtop);
    attn_update_kernel<<<dim3(CB * CH * CU), 256, 0, stream>>>(qbuf, kbuf, vbuf,
                                                               mtop, upd);
    vmean_kernel<<<dim3(CB * CH), 64, 0, stream>>>(vbuf, vmean);
    ctx_fill_kernel<<<dim3((unsigned)(((size_t)CM * CDIM + 255) / 256)), 256, 0,
                      stream>>>(vmean, ctx);
    ctx_scatter_kernel<<<dim3(CB * CH * CU), 64, 0, stream>>>(mtop, upd, ctx);

    cvt(ctx, act16, (size_t)CM * CDIM);
    gemm(act16, wo16, bo, tmp, CM, CDIM, CDIM);
  };

  // ctx_b: q from after_norm, k/v from before_norm
  attention(na, nb);
  add_ln_kernel<<<dim3(CM), 256, 0, stream>>>(nb, tmp, norm_w, norm_b, bstate, CDIM);
  // ctx_a: q from before_norm, k/v from after_norm
  attention(nb, na);
  add_ln_kernel<<<dim3(CM), 256, 0, stream>>>(na, tmp, norm_w, norm_b, astate, CDIM);

  // ---- MLP: out = x + LN2(fc2(GELU(LN1(fc1(x))))) -------------------------
  auto mlp = [&](const float* x, float* outp) {
    cvt(x, act16, (size_t)CM * CDIM);
    gemm(act16, fc1w16, fc1_b, hid, CM, CHID, CDIM);
    ln(hid, ln1_w, ln1_b, hid, CHID, /*gelu=*/1);
    cvt(hid, act16, (size_t)CM * CHID);
    gemm(act16, fc2w16, fc2_b, tmp, CM, CDIM, CHID);
    ln(tmp, ln2_w, ln2_b, tmp, CDIM, 0);
    add_kernel<<<dim3((unsigned)(((size_t)CM * CDIM + 255) / 256)), 256, 0,
                 stream>>>(x, tmp, outp, (size_t)CM * CDIM);
  };

  mlp(bstate, out);                               // stacked output [0]
  mlp(astate, out + (size_t)CM * CDIM);           // stacked output [1]
}
